// GTLUTProduct_72748156060365
// MI455X (gfx1250) — compile-verified, matrix operand order checked by reading, not run on hardware
//
#include <hip/hip_runtime.h>
#include <hip/hip_bf16.h>

// Problem constants (from the reference)
#define BB     4
#define LL     320
#define N1     48
#define N2     48
#define PD     48
#define N_OUT  64
#define N_DET  8
#define N_ANCH 6
#define N_LUT_IN  (N1 + N2 + PD)   // 144
#define N_CODES   64
#define KDIM      (N_DET * N_CODES) // 512
#define NTOK      (BB * LL)         // 1280
#define NGROUP    (NTOK / 16)       // 80 groups of 16 tokens (M-tiles)
#define KSTEPS    (KDIM / 32)       // 16 WMMA K-steps
#define NTILES    (N_OUT / 16)      // 4 N-tiles

typedef __attribute__((ext_vector_type(16))) _Float16 v16h;
typedef __attribute__((ext_vector_type(8)))  float    v8f;

// -------- WMMA 16-bit fragment index helpers (ISA 7.12.2 layouts) --------
// For A (16x32 f16) and B (32x16 f16): lane half h = lane/16,
// element e in v16h maps to K = (e<8 ? e : e+8) + 8*h.
__device__ __forceinline__ int kk_of_lane_elem(int lane, int e) {
    return (e & 7) + ((e >> 3) << 4) + ((lane >> 4) << 3);
}

// -------- Kernel 0: repack lut_weights (f32 [8][64][64]) into f16 B-fragments
// Wfrag layout: [nt][kstep][lane][e], 4*16*32*16 halves = 64 KB.
__global__ void wfrag_kernel(const float* __restrict__ w, _Float16* __restrict__ wfrag) {
    int idx = blockIdx.x * blockDim.x + threadIdx.x;     // 0..32767
    int e    = idx & 15;
    int lane = (idx >> 4) & 31;
    int ks   = (idx >> 9) & 15;
    // nt = idx >> 13 (implicit in o below)
    int nt   = idx >> 13;
    int kk   = kk_of_lane_elem(lane, e);
    int k    = ks * 32 + kk;           // global K index 0..511
    int d    = k >> 6;                 // detector
    int c    = k & 63;                 // code
    int n    = lane & 15;
    int o    = nt * 16 + n;            // output channel
    wfrag[idx] = (_Float16)w[(d * N_CODES + c) * N_OUT + o];
}

// -------- Kernel 1: accumulate probs over i<j, store as A-fragments (f16)
__device__ __forceinline__ float triple_val(int idx, int b, int i, int j,
                                            const float* __restrict__ in1,
                                            const float* __restrict__ in2,
                                            const float* __restrict__ pe) {
    if (idx < N1)        return in1[(b * LL + i) * N1 + idx];
    else if (idx < N1+N2) return in2[(b * LL + j) * N2 + (idx - N1)];
    else                  return pe[(j - i - 1) * PD + (idx - N1 - N2)];
}

__global__ void __launch_bounds__(256)
lut_probs_kernel(const float* __restrict__ in1, const float* __restrict__ in2,
                 const float* __restrict__ pe,
                 const int* __restrict__ aa, const int* __restrict__ ab,
                 _Float16* __restrict__ afrag) {
    __shared__ float p_lds[32][48];   // sigmoid(diff) per i-chunk, per (d,k)
    __shared__ int a_s[48], b_s[48];

    int tid = threadIdx.x;
    int bj  = blockIdx.x;             // token index = b*LL + j
    int b   = bj / LL;
    int j   = bj - b * LL;

    if (tid < 48) { a_s[tid] = aa[tid]; b_s[tid] = ab[tid]; }
    __syncthreads();

    // Each thread owns two (d,c) accumulators: dc0 = tid, dc1 = tid+256.
    int dc0 = tid, dc1 = tid + 256;
    int d0 = dc0 >> 6, c0 = dc0 & 63;
    int d1 = dc1 >> 6, c1 = dc1 & 63;
    float acc0 = 0.0f, acc1 = 0.0f;

    for (int i0 = 0; i0 < j; i0 += 32) {
        int nI = j - i0; if (nI > 32) nI = 32;
        int ntask = nI * 48;
        for (int task = tid; task < ntask; task += 256) {
            int il = task / 48;
            int dk = task - il * 48;
            int i  = i0 + il;
            float va = triple_val(a_s[dk], b, i, j, in1, in2, pe);
            float vb = triple_val(b_s[dk], b, i, j, in1, in2, pe);
            float diff = va - vb;
            p_lds[il][dk] = 1.0f / (1.0f + __expf(-diff));  // sigmoid
        }
        __syncthreads();

        for (int il = 0; il < nI; ++il) {
            const float* p0 = &p_lds[il][d0 * N_ANCH];
            const float* p1 = &p_lds[il][d1 * N_ANCH];
            float pr0 = 1.0f, pr1 = 1.0f;
            #pragma unroll
            for (int k = 0; k < N_ANCH; ++k) {
                float q0 = p0[k];
                float q1 = p1[k];
                pr0 *= ((c0 >> k) & 1) ? q0 : (1.0f - q0);
                pr1 *= ((c1 >> k) & 1) ? q1 : (1.0f - q1);
            }
            acc0 += pr0;
            acc1 += pr1;
        }
        __syncthreads();
    }

    // Store into WMMA A-fragment layout: Afrag[group][kstep][lane][e]
    int group = bj >> 4;
    int m     = bj & 15;
    {
        int ks = dc0 >> 5, kk = dc0 & 31;
        int e    = (kk & 7) + ((kk >> 4) << 3);
        int lane = (((kk >> 3) & 1) << 4) + m;
        afrag[(((group * KSTEPS) + ks) * 32 + lane) * 16 + e] = (_Float16)acc0;
    }
    {
        int ks = dc1 >> 5, kk = dc1 & 31;
        int e    = (kk & 7) + ((kk >> 4) << 3);
        int lane = (((kk >> 3) & 1) << 4) + m;
        afrag[(((group * KSTEPS) + ks) * 32 + lane) * 16 + e] = (_Float16)acc1;
    }
}

// -------- Kernel 2: per 16-token group, S(16x512) x W(512x64) via WMMA f16
__global__ void __launch_bounds__(32)
lut_gemm_kernel(const _Float16* __restrict__ afrag, const _Float16* __restrict__ wfrag,
                float* __restrict__ out) {
    int group = blockIdx.x;           // 0..79
    int lane  = threadIdx.x;          // 0..31 (wave32)

    v8f acc[NTILES];
    #pragma unroll
    for (int nt = 0; nt < NTILES; ++nt) acc[nt] = (v8f){0,0,0,0,0,0,0,0};

    #pragma unroll 4
    for (int ks = 0; ks < KSTEPS; ++ks) {
        v16h a = *(const v16h*)(afrag + (((group * KSTEPS) + ks) * 32 + lane) * 16);
        #pragma unroll
        for (int nt = 0; nt < NTILES; ++nt) {
            v16h bf = *(const v16h*)(wfrag + (((nt * KSTEPS) + ks) * 32 + lane) * 16);
            acc[nt] = __builtin_amdgcn_wmma_f32_16x16x32_f16(
                /*neg_a=*/false, a, /*neg_b=*/false, bf,
                /*c_mod=*/(short)0, acc[nt], /*reuse_a=*/false, /*reuse_b=*/false);
        }
    }

    // C/D layout: VGPR r -> M = r + 8*(lane/16), N = lane%16
    int n  = lane & 15;
    int hs = lane >> 4;
    #pragma unroll
    for (int nt = 0; nt < NTILES; ++nt) {
        #pragma unroll
        for (int r = 0; r < 8; ++r) {
            int token = group * 16 + (r + 8 * hs);
            out[token * N_OUT + nt * 16 + n] = acc[nt][r];
        }
    }
}

extern "C" void kernel_launch(void* const* d_in, const int* in_sizes, int n_in,
                              void* d_out, int out_size, void* d_ws, size_t ws_size,
                              hipStream_t stream) {
    const float* input_1   = (const float*)d_in[0];   // (B, L, 48)
    const float* input_2   = (const float*)d_in[1];   // (B, L, 48)
    const float* pos_emb   = (const float*)d_in[2];   // (L-1, 48)
    const float* lut_w     = (const float*)d_in[3];   // (8, 64, 64)
    const int*   anchors_a = (const int*)d_in[4];     // (8, 6)
    const int*   anchors_b = (const int*)d_in[5];     // (8, 6)
    float*       out       = (float*)d_out;           // (B, L, 64)

    // Workspace layout: Wfrag (64 KB f16) then Afrag (1.25 MB f16)
    _Float16* wfrag = (_Float16*)d_ws;
    _Float16* afrag = (_Float16*)((char*)d_ws + (size_t)NTILES * KSTEPS * 32 * 16 * sizeof(_Float16));

    // 1) Repack weights into WMMA B-fragments (32768 elements)
    wfrag_kernel<<<(NTILES * KSTEPS * 32 * 16) / 256, 256, 0, stream>>>(lut_w, wfrag);

    // 2) Accumulate probabilities per token into WMMA A-fragments
    lut_probs_kernel<<<NTOK, 256, 0, stream>>>(input_1, input_2, pos_emb,
                                               anchors_a, anchors_b, afrag);

    // 3) WMMA GEMM: out[16tok x 64] = S[16 x 512] * W[512 x 64] per group
    lut_gemm_kernel<<<NGROUP, 32, 0, stream>>>(afrag, wfrag, out);
}
